// Attention_46712064312136
// MI455X (gfx1250) — compile-verified, matrix operand order checked by reading, not run on hardware
//
#include <hip/hip_runtime.h>
#include <hip/hip_bf16.h>

#define B_  2
#define S_  2048
#define DIM_ 1024
#define NH_ 16
#define NKV_ 8
#define HD_ 128
#define EPS_ 1e-6f
#define GEMM_PAD 40   // LDS row stride in bf16 elems (80B) -> conflict-free rows

typedef __bf16 bf16_t;
typedef __attribute__((ext_vector_type(16))) __bf16 v16bf;
typedef __attribute__((ext_vector_type(8)))  __bf16 v8bf;
typedef __attribute__((ext_vector_type(4)))  __bf16 v4bf;
typedef __attribute__((ext_vector_type(8)))  float  v8f;

// ---------------------------------------------------------------------------
// 16-bit A/B fragment loader from row-major storage (16 rows x 32 K).
// Lane L (r = L&15, kg = L>>4) holds K = {kg*8+0..7} and {16+kg*8+0..7}.
// Two 16-byte loads per lane (global_load_b128 or ds_load_b128).
// ---------------------------------------------------------------------------
__device__ __forceinline__ v16bf load_frag16(const bf16_t* __restrict__ base,
                                             int row, int ld, int k0) {
  const int lane = threadIdx.x & 31;
  const int r    = lane & 15;
  const int kg   = lane >> 4;
  const bf16_t* p = base + (size_t)(row + r) * (size_t)ld + k0 + kg * 8;
  v8bf lo = *reinterpret_cast<const v8bf*>(p);
  v8bf hi = *reinterpret_cast<const v8bf*>(p + 16);
  v16bf f;
#pragma unroll
  for (int i = 0; i < 8; ++i) { f[i] = lo[i]; f[8 + i] = hi[i]; }
  return f;
}

// ---------------------------------------------------------------------------
// f32 -> bf16 conversion (4 elements / thread, vectorized)
// ---------------------------------------------------------------------------
__global__ void f32_to_bf16_kernel(const float* __restrict__ in,
                                   bf16_t* __restrict__ out, int n) {
  int i = (blockIdx.x * blockDim.x + threadIdx.x) * 4;
  if (i < n) {
    float4 v = *reinterpret_cast<const float4*>(in + i);
    v4bf o;
    o[0] = (bf16_t)v.x; o[1] = (bf16_t)v.y; o[2] = (bf16_t)v.z; o[3] = (bf16_t)v.w;
    *reinterpret_cast<v4bf*>(out + i) = o;
  }
}

// ---------------------------------------------------------------------------
// GEMM: C[M,N] f32 = A[M,K] bf16 (row-major) x W[N,K] bf16 (row-major = B^T).
// Block = 128 threads (4 waves), C tile 64x64.  The 64x32 W slab (shared by
// all 4 waves) is staged in LDS with GLOBAL_LOAD_ASYNC_TO_LDS_B128 (ASYNCcnt),
// double-buffered so the async DMA overlaps the WMMA stream; fragments are
// then read back with ds_load_b128.  A fragments stream from global (L2-hot).
// grid = (N/64, M/64).
// ---------------------------------------------------------------------------
__global__ __launch_bounds__(128) void gemm_bf16_wmma(
    const bf16_t* __restrict__ A, const bf16_t* __restrict__ W,
    float* __restrict__ C, int M, int N, int K) {
  __shared__ alignas(16) bf16_t wt[2][64 * GEMM_PAD];
  const int tid  = threadIdx.x;
  const int lane = tid & 31;
  const int wave = tid >> 5;
  const int m0 = blockIdx.y * 64 + wave * 16;
  const int n0 = blockIdx.x * 64;

  v8f acc[4];
#pragma unroll
  for (int t = 0; t < 4; ++t)
#pragma unroll
    for (int i = 0; i < 8; ++i) acc[t][i] = 0.0f;

  // Stage W rows [n0, n0+64) x K-slab [k0, k0+32) into wt[buf]:
  // 256 16-byte chunks, 2 per thread, via async copy (no VGPR data path).
  auto stage = [&](int buf, int k0) {
#pragma unroll
    for (int c = 0; c < 2; ++c) {
      const int ch  = tid * 2 + c;      // 0..255
      const int row = ch >> 2;          // 0..63
      const int seg = ch & 3;           // 4 x 16B per 64B row
      const bf16_t* gp = W + (size_t)(n0 + row) * K + k0 + seg * 8;
      const unsigned lo =
          (unsigned)(size_t)&wt[buf][row * GEMM_PAD + seg * 8];
      asm volatile("global_load_async_to_lds_b128 %0, %1, off"
                   :: "v"(lo), "v"(gp) : "memory");
    }
  };

  stage(0, 0);
  asm volatile("s_wait_asynccnt 0x0" ::: "memory");
  __syncthreads();

  for (int k0 = 0; k0 < K; k0 += 32) {
    const int cur = (k0 >> 5) & 1;
    if (k0 + 32 < K) {
      stage(cur ^ 1, k0 + 32);          // async prefetch of next W slab
      __builtin_prefetch(A + (size_t)(m0 + (lane & 15)) * K + k0 + 32, 0, 0);
    }
    v16bf a = load_frag16(A, m0, K, k0);
#pragma unroll
    for (int t = 0; t < 4; ++t) {
      v16bf b = load_frag16(&wt[cur][16 * t * GEMM_PAD], 0, GEMM_PAD, 0);
      acc[t] = __builtin_amdgcn_wmma_f32_16x16x32_bf16(
          false, a, false, b, (short)0, acc[t], false, false);
    }
    asm volatile("s_wait_asynccnt 0x0" ::: "memory");
    __syncthreads();                    // next slab ready, this one consumed
  }
  const int r16 = lane & 15;
  const int hi8 = (lane >> 4) * 8;
#pragma unroll
  for (int t = 0; t < 4; ++t)
#pragma unroll
    for (int r = 0; r < 8; ++r)
      C[(size_t)(m0 + hi8 + r) * N + n0 + 16 * t + r16] = acc[t][r];
}

// ---------------------------------------------------------------------------
// Fused per-head RMSNorm + RoPE + bf16 pack + layout shuffle.
// One wave per (b, s, head-slot); slots 0..15 = Q heads, 16..23 = K heads,
// 24..31 = V heads.  Lane owns dims d = lane*4 .. lane*4+3; rotate-half
// partner (d ^ 64) is lane ^ 16.
// Outputs: Qb (B,NH,S,HD), Kb (B,NKV,S,HD), Vt (B,NKV,HD,S)  (all bf16).
// ---------------------------------------------------------------------------
__global__ __launch_bounds__(128) void normrope_kernel(
    const float* __restrict__ Qf, const float* __restrict__ Kf,
    const float* __restrict__ Vf, const float* __restrict__ rope,
    const float* __restrict__ qw, const float* __restrict__ kw,
    bf16_t* __restrict__ Qb, bf16_t* __restrict__ Kb, bf16_t* __restrict__ Vt) {
  const int wid  = blockIdx.x * 4 + (threadIdx.x >> 5);
  const int lane = threadIdx.x & 31;
  const int bs  = wid >> 5;      // (b * S + s)
  const int hid = wid & 31;      // head slot
  const int b = bs / S_;
  const int s = bs - b * S_;
  const int d0 = lane * 4;
  const float* ropeRow = rope + (size_t)s * (2 * HD_);

  if (hid < NH_ + NKV_) {  // Q or K: RMSNorm + RoPE
    const bool isQ = hid < NH_;
    const int h = isQ ? hid : hid - NH_;
    const float* src = isQ ? (Qf + (size_t)bs * (NH_ * HD_) + h * HD_)
                           : (Kf + (size_t)bs * (NKV_ * HD_) + h * HD_);
    const float* w = isQ ? qw : kw;
    float4 t = *reinterpret_cast<const float4*>(src + d0);
    float ss = t.x * t.x + t.y * t.y + t.z * t.z + t.w * t.w;
#pragma unroll
    for (int off = 16; off > 0; off >>= 1) ss += __shfl_xor(ss, off, 32);
    const float inv = rsqrtf(ss * (1.0f / HD_) + EPS_);
    float nv[4] = { t.x * inv * w[d0], t.y * inv * w[d0 + 1],
                    t.z * inv * w[d0 + 2], t.w * inv * w[d0 + 3] };
    float pv[4];
#pragma unroll
    for (int i = 0; i < 4; ++i) pv[i] = __shfl_xor(nv[i], 16, 32);
    const float sgn = (lane < 16) ? -1.0f : 1.0f;  // rotate_half sign
    v4bf ov;
#pragma unroll
    for (int i = 0; i < 4; ++i) {
      const float c  = ropeRow[d0 + i];
      const float si = ropeRow[HD_ + d0 + i];
      ov[i] = (bf16_t)(nv[i] * c + sgn * pv[i] * si);
    }
    bf16_t* dst = isQ ? (Qb + ((size_t)(b * NH_ + h) * S_ + s) * HD_)
                      : (Kb + ((size_t)(b * NKV_ + h) * S_ + s) * HD_);
    *reinterpret_cast<v4bf*>(dst + d0) = ov;
  } else {                 // V: plain bf16 convert + transpose to (HD, S)
    const int h = hid - (NH_ + NKV_);
    const float* src = Vf + (size_t)bs * (NKV_ * HD_) + h * HD_;
    float4 t = *reinterpret_cast<const float4*>(src + d0);
    bf16_t* dst = Vt + (size_t)(b * NKV_ + h) * HD_ * S_ + s;
    dst[(size_t)(d0 + 0) * S_] = (bf16_t)t.x;
    dst[(size_t)(d0 + 1) * S_] = (bf16_t)t.y;
    dst[(size_t)(d0 + 2) * S_] = (bf16_t)t.z;
    dst[(size_t)(d0 + 3) * S_] = (bf16_t)t.w;
  }
}

// ---------------------------------------------------------------------------
// Causal flash attention, one wave per 16-query tile.
// Scores computed transposed (S^T = K_tile x Q^T) so the per-query softmax is
// an in-register reduction + one shfl_xor(16), and exp'd probs repack into
// the PV WMMA's B-fragment entirely in-lane.  O^T = V^T x P^T, 8 f32 frags.
// grid = (S/64, NH, B), block = 128 (4 waves, 4 query tiles).
// ---------------------------------------------------------------------------
__global__ __launch_bounds__(128) void flash_attn_wmma(
    const bf16_t* __restrict__ Qb, const bf16_t* __restrict__ Kb,
    const bf16_t* __restrict__ Vt, bf16_t* __restrict__ AO) {
  const int lane = threadIdx.x & 31;
  const int wave = threadIdx.x >> 5;
  const int b = blockIdx.z, h = blockIdx.y;
  const int kv = h >> 1;                       // N_REP = 2
  const int q0 = blockIdx.x * 64 + wave * 16;
  const bf16_t* Qp = Qb + (size_t)(b * NH_ + h) * S_ * HD_;
  const bf16_t* Kp = Kb + (size_t)(b * NKV_ + kv) * S_ * HD_;
  const bf16_t* Vp = Vt + (size_t)(b * NKV_ + kv) * HD_ * S_;

  v16bf bq[4];                                 // resident Q B-fragments (K=HD)
#pragma unroll
  for (int j = 0; j < 4; ++j) bq[j] = load_frag16(Qp, q0, HD_, j * 32);

  v8f o[8];                                    // O^T accumulators (HD x 16)
#pragma unroll
  for (int t = 0; t < 8; ++t)
#pragma unroll
    for (int i = 0; i < 8; ++i) o[t][i] = 0.0f;

  float m_run = -1e30f, l_run = 0.0f;
  const float scale = 0.088388347648318447f;   // 1/sqrt(128)
  const int n   = lane & 15;                   // this lane's query column
  const int hi8 = (lane >> 4) * 8;
  const int qg  = q0 + n;
  const int nchunks = (q0 + 16 + 31) >> 5;     // 32 keys per chunk

  for (int c = 0; c < nchunks; ++c) {
    const int kb = c * 32;
    v8f st0, st1;
#pragma unroll
    for (int i = 0; i < 8; ++i) { st0[i] = 0.0f; st1[i] = 0.0f; }
#pragma unroll
    for (int j = 0; j < 4; ++j) {              // S^T = K x Q^T over HD=128
      v16bf ak0 = load_frag16(Kp, kb, HD_, j * 32);
      st0 = __builtin_amdgcn_wmma_f32_16x16x32_bf16(
          false, ak0, false, bq[j], (short)0, st0, false, false);
      v16bf ak1 = load_frag16(Kp, kb + 16, HD_, j * 32);
      st1 = __builtin_amdgcn_wmma_f32_16x16x32_bf16(
          false, ak1, false, bq[j], (short)0, st1, false, false);
    }
    float s0[8], s1[8];
    float cmax = -1e30f;
#pragma unroll
    for (int r = 0; r < 8; ++r) {              // scale + causal mask (no branch)
      const int kg0 = kb + hi8 + r;
      s0[r] = (kg0      <= qg) ? st0[r] * scale : -1e30f;
      s1[r] = (kg0 + 16 <= qg) ? st1[r] * scale : -1e30f;
      cmax = fmaxf(cmax, fmaxf(s0[r], s1[r]));
    }
    cmax = fmaxf(cmax, __shfl_xor(cmax, 16, 32));
    const float m_new = fmaxf(m_run, cmax);
    const float fac   = __expf(m_run - m_new);
    float rs = 0.0f;
    v16bf pb;                                  // P^T B-fragment, built in-lane
#pragma unroll
    for (int r = 0; r < 8; ++r) {
      const float p0 = __expf(s0[r] - m_new);
      const float p1 = __expf(s1[r] - m_new);
      rs += p0 + p1;
      pb[r]     = (bf16_t)p0;
      pb[8 + r] = (bf16_t)p1;
    }
    rs += __shfl_xor(rs, 16, 32);
    l_run = l_run * fac + rs;
    m_run = m_new;
#pragma unroll
    for (int t = 0; t < 8; ++t)
#pragma unroll
      for (int i = 0; i < 8; ++i) o[t][i] *= fac;
#pragma unroll
    for (int t = 0; t < 8; ++t) {              // O^T += V^T x P^T
      v16bf av = load_frag16(Vp, t * 16, S_, kb);
      o[t] = __builtin_amdgcn_wmma_f32_16x16x32_bf16(
          false, av, false, pb, (short)0, o[t], false, false);
    }
  }
  const float invl = 1.0f / l_run;
  bf16_t* dst = AO + ((size_t)(b * S_ + q0 + n) * NH_ + h) * HD_;
#pragma unroll
  for (int t = 0; t < 8; ++t)
#pragma unroll
    for (int r = 0; r < 8; ++r)
      dst[t * 16 + hi8 + r] = (bf16_t)(o[t][r] * invl);
}

// ---------------------------------------------------------------------------
extern "C" void kernel_launch(void* const* d_in, const int* in_sizes, int n_in,
                              void* d_out, int out_size, void* d_ws, size_t ws_size,
                              hipStream_t stream) {
  const float* x    = (const float*)d_in[0];
  const float* rope = (const float*)d_in[1];
  const float* wq   = (const float*)d_in[2];
  const float* wk   = (const float*)d_in[3];
  const float* wv   = (const float*)d_in[4];
  const float* wo   = (const float*)d_in[5];
  const float* qnw  = (const float*)d_in[6];
  const float* knw  = (const float*)d_in[7];

  const int BS = B_ * S_;                       // 4096 rows
  char* ws = (char*)d_ws;
  size_t off = 0;
  auto take = [&](size_t bytes) {
    char* p = ws + off;
    off = (off + bytes + 255) & ~(size_t)255;
    return p;
  };
  bf16_t* xb  = (bf16_t*)take((size_t)BS * DIM_ * 2);
  bf16_t* wqb = (bf16_t*)take((size_t)NH_ * HD_ * DIM_ * 2);
  bf16_t* wkb = (bf16_t*)take((size_t)NKV_ * HD_ * DIM_ * 2);
  bf16_t* wvb = (bf16_t*)take((size_t)NKV_ * HD_ * DIM_ * 2);
  bf16_t* wob = (bf16_t*)take((size_t)DIM_ * NH_ * HD_ * 2);
  float*  Qf  = (float*) take((size_t)BS * NH_ * HD_ * 4);
  float*  Kf  = (float*) take((size_t)BS * NKV_ * HD_ * 4);
  float*  Vf  = (float*) take((size_t)BS * NKV_ * HD_ * 4);
  bf16_t* Qb  = (bf16_t*)take((size_t)BS * NH_ * HD_ * 2);
  bf16_t* Kb  = (bf16_t*)take((size_t)BS * NKV_ * HD_ * 2);
  bf16_t* Vt  = (bf16_t*)take((size_t)BS * NKV_ * HD_ * 2);
  bf16_t* AO  = (bf16_t*)take((size_t)BS * NH_ * HD_ * 2);

  auto cvt = [&](const float* src, bf16_t* dst, int n) {
    f32_to_bf16_kernel<<<(n / 4 + 255) / 256, 256, 0, stream>>>(src, dst, n);
  };
  cvt(x,  xb,  BS * DIM_);
  cvt(wq, wqb, NH_ * HD_ * DIM_);
  cvt(wk, wkb, NKV_ * HD_ * DIM_);
  cvt(wv, wvb, NKV_ * HD_ * DIM_);
  cvt(wo, wob, DIM_ * NH_ * HD_);

  // QKV projections (block computes 64x64, W slab async-staged in LDS)
  gemm_bf16_wmma<<<dim3((NH_ * HD_) / 64, BS / 64), 128, 0, stream>>>(
      xb, wqb, Qf, BS, NH_ * HD_, DIM_);
  gemm_bf16_wmma<<<dim3((NKV_ * HD_) / 64, BS / 64), 128, 0, stream>>>(
      xb, wkb, Kf, BS, NKV_ * HD_, DIM_);
  gemm_bf16_wmma<<<dim3((NKV_ * HD_) / 64, BS / 64), 128, 0, stream>>>(
      xb, wvb, Vf, BS, NKV_ * HD_, DIM_);

  // RMSNorm + RoPE + layout
  normrope_kernel<<<(BS * (NH_ + 2 * NKV_)) / 4, 128, 0, stream>>>(
      Qf, Kf, Vf, rope, qnw, knw, Qb, Kb, Vt);

  // Causal flash attention
  flash_attn_wmma<<<dim3(S_ / 64, NH_, B_), 128, 0, stream>>>(Qb, Kb, Vt, AO);

  // Output projection -> d_out (f32)
  gemm_bf16_wmma<<<dim3(DIM_ / 64, BS / 64), 128, 0, stream>>>(
      AO, wob, (float*)d_out, BS, DIM_, NH_ * HD_);
}
